// SingleresCNN_72962904425054
// MI455X (gfx1250) — matrix operand order repaired and checked by reading, not compiled, not run on hardware
//
#include <hip/hip_runtime.h>

// ---------------------------------------------------------------------------
// Types for CDNA5 WMMA (wave32, V_WMMA_F32_16X16X32_F16)
// ---------------------------------------------------------------------------
typedef __attribute__((ext_vector_type(16))) _Float16 v16h;
typedef __attribute__((ext_vector_type(8)))  _Float16 half8;
typedef __attribute__((ext_vector_type(4)))  _Float16 half4;
typedef __attribute__((ext_vector_type(8)))  float    v8f;
typedef __attribute__((ext_vector_type(4)))  float    v4f;

#define NPIX 524288   // 8 * 256 * 256
#define HW   65536    // 256 * 256
#define WIMG 256

union V16U { v16h v; half8 h[2]; };

__device__ __forceinline__ half8 h8_zero() {
  half8 z = { (_Float16)0.0f, (_Float16)0.0f, (_Float16)0.0f, (_Float16)0.0f,
              (_Float16)0.0f, (_Float16)0.0f, (_Float16)0.0f, (_Float16)0.0f };
  return z;
}

__device__ __forceinline__ v8f wmma_f16(v16h a, v16h b, v8f c) {
  // (neg_a, A, neg_b, B, c_mod, C, reuse_a, reuse_b)
  return __builtin_amdgcn_wmma_f32_16x16x32_f16(false, a, false, b, (short)0, c,
                                                false, false);
}

// ---------------------------------------------------------------------------
// Weight pre-pack: OIHW f32 -> WMMA-B-layout f16 blocks of 512 halfs.
// Block index = ((t*nkc + kc)*nocg + og); within a block element (lane, i):
//   column n = lane & 15, K = (lane<16 ? i : 16+i), ci = kc*32 + K
// ---------------------------------------------------------------------------
__global__ void pack_w(const float* __restrict__ W, _Float16* __restrict__ dst,
                       int cinReal, int nkc, int nocg, int taps) {
  int idx = blockIdx.x * blockDim.x + threadIdx.x;
  int total = taps * nkc * nocg * 512;
  if (idx >= total) return;
  int i    = idx & 15;
  int lane = (idx >> 4) & 31;
  int blk  = idx >> 9;
  int og   = blk % nocg;
  int tkc  = blk / nocg;
  int kc   = tkc % nkc;
  int t    = tkc / nkc;
  int n    = lane & 15;
  int kk   = (lane < 16) ? i : (16 + i);
  int ci   = (kc << 5) + kk;
  int oc   = (og << 4) + n;
  float w = (ci < cinReal) ? W[((size_t)oc * cinReal + ci) * taps + t] : 0.0f;
  dst[idx] = (_Float16)w;
}

// ---------------------------------------------------------------------------
// feat(NCHW f32) * mask -> NHWC f16 activation (C = 32)
// ---------------------------------------------------------------------------
__global__ void prep_feat(const float* __restrict__ feat,
                          const float* __restrict__ mask,
                          _Float16* __restrict__ act) {
  size_t idx = (size_t)blockIdx.x * blockDim.x + threadIdx.x;  // grid exact
  int c   = (int)(idx / NPIX);
  int pix = (int)(idx - (size_t)c * NPIX);
  int b   = pix >> 16;
  int yx  = pix & (HW - 1);
  float v = feat[((size_t)b * 32 + c) * HW + yx] * mask[pix];
  act[(size_t)pix * 32 + c] = (_Float16)v;
}

// ---------------------------------------------------------------------------
// Mask step: 3x3 dilated box-sum (for partial-conv count) + maxpool (next mask)
// Zero padding outside the image (matches jnp.pad + VALID windows).
// ---------------------------------------------------------------------------
__global__ void mask_step(const float* __restrict__ m, float* __restrict__ boxsum,
                          float* __restrict__ mnext, int d) {
  int pix = blockIdx.x * blockDim.x + threadIdx.x;
  if (pix >= NPIX) return;
  int b = pix >> 16, yx = pix & (HW - 1);
  int y = yx >> 8,  x  = yx & (WIMG - 1);
  const float* mb = m + (size_t)b * HW;
  float s = 0.0f, mx = 0.0f;
#pragma unroll
  for (int ky = 0; ky < 3; ++ky)
#pragma unroll
    for (int kx = 0; kx < 3; ++kx) {
      int iy = y + (ky - 1) * d, ix = x + (kx - 1) * d;
      float v = 0.0f;
      if (iy >= 0 && iy < WIMG && ix >= 0 && ix < WIMG) v = mb[iy * WIMG + ix];
      s += v;
      mx = fmaxf(mx, v);
    }
  boxsum[pix] = s;
  mnext[pix]  = mx;
}

// ---------------------------------------------------------------------------
// 3x3 dilated sparse conv, implicit GEMM via WMMA f16 -> f32.
// Block = 8 waves sharing ONE out-channel group over 128 contiguous pixels;
// that group's packed weights (9*NKC*512 halfs) are staged in LDS once and
// consumed as WMMA B operands via ds_load. A operands are loaded straight
// from NHWC global memory with clamped addresses + cndmask zeroing so EXEC
// stays all-ones through the WMMA stream.
// y = conv(act)/max(boxsum*cin, 1e-5) + bias, stored NHWC f32 (padded CpOut).
// ---------------------------------------------------------------------------
template <int NKC>
__global__ __launch_bounds__(256) void conv3x3_sparse_wmma(
    const _Float16* __restrict__ act, const _Float16* __restrict__ wp,
    const float* __restrict__ boxsum, const float* __restrict__ bias,
    float* __restrict__ ybuf, int CpIn, int nocg, int CpOut, int dil,
    float cinReal) {
  __shared__ alignas(32) _Float16 wlds[9 * NKC * 512];

  const int og    = blockIdx.x % nocg;
  const int tile8 = blockIdx.x / nocg;

  // Cooperative global->LDS stage of this og's weight slice (16B per thread).
  const int totalH = 9 * NKC * 512;
  for (int idx = threadIdx.x * 8; idx < totalH; idx += 256 * 8) {
    const int blk    = idx >> 9;
    const int within = idx & 511;
    *(half8*)&wlds[idx] =
        *(const half8*)(wp + ((size_t)(blk * nocg + og) << 9) + within);
  }
  __syncthreads();

  const int lane    = threadIdx.x & 31;
  const int pixbase = (tile8 << 7) + ((threadIdx.x >> 5) << 4);
  const int b  = pixbase >> 16;
  const int yx = pixbase & (HW - 1);
  const int yy = yx >> 8;
  const int x0 = yx & (WIMG - 1);
  const int pl = lane & 15;   // A row / D column index
  const int hi = lane >> 4;   // K-half / M-half selector

  __builtin_prefetch(boxsum + pixbase + (hi << 3), 0, 0);  // epilogue data

  v8f acc = {0.f, 0.f, 0.f, 0.f, 0.f, 0.f, 0.f, 0.f};

  for (int t = 0; t < 9; ++t) {
    const int kh = t / 3, kw = t % 3;
    const int iy = yy + (kh - 1) * dil;
    if (iy < 0 || iy >= WIMG) continue;  // block-uniform: whole tap is zero-pad
    const int ix  = x0 + pl + (kw - 1) * dil;
    const bool vx = (ix >= 0) && (ix < WIMG);
    const int ixc = vx ? ix : 0;         // clamp -> always-valid address
    const _Float16* arow =
        act + (size_t)((b * WIMG + iy) * WIMG + ixc) * CpIn;
#pragma unroll
    for (int kc = 0; kc < NKC; ++kc) {
      const int c0 = (kc << 5) + (hi << 4);
      half8 l0 = *(const half8*)(arow + c0);        // K run (8 halfs, 16B)
      half8 l1 = *(const half8*)(arow + c0 + 8);    // K run +8
      V16U a;
      a.h[0] = vx ? l0 : h8_zero();
      a.h[1] = vx ? l1 : h8_zero();
      const v16h bv = *(const v16h*)&wlds[((t * NKC + kc) << 9) + (lane << 4)];
      acc = wmma_f16(a.v, bv, acc);
    }
  }

  // Count-normalize + bias, store y (NHWC, padded CpOut stride).
  const float* bsp = boxsum + pixbase + (hi << 3);
  v4f bs0 = *(const v4f*)bsp;
  v4f bs1 = *(const v4f*)(bsp + 4);
  const float bc = bias[(og << 4) + pl];
  float* yo = ybuf + (size_t)(pixbase + (hi << 3)) * CpOut + (og << 4) + pl;
#pragma unroll
  for (int r = 0; r < 8; ++r) {
    float bs  = (r < 4) ? bs0[r] : bs1[r - 4];
    float cnt = fmaxf(bs * cinReal, 1e-5f);
    yo[(size_t)r * CpOut] = acc[r] / cnt + bc;
  }
}

// ---------------------------------------------------------------------------
// BatchNorm stats: per-channel sum / sumsq over all 524288 pixels.
// C in {16,32,64,128} divides 256 and is a power of two.
// ---------------------------------------------------------------------------
__global__ __launch_bounds__(256) void bn_stats(const float* __restrict__ y,
                                                float* __restrict__ sum,
                                                float* __restrict__ sumsq,
                                                int C, int Cp) {
  __shared__ float s[256], q[256];
  const int tid  = threadIdx.x;
  const int c    = tid & (C - 1);
  const int row0 = tid / C;
  const int rpb  = 256 / C;
  float a = 0.0f, b2 = 0.0f;
  for (int p = blockIdx.x * rpb + row0; p < NPIX; p += gridDim.x * rpb) {
    float v = y[(size_t)p * Cp + c];
    a  += v;
    b2 += v * v;
  }
  s[tid] = a;
  q[tid] = b2;
  __syncthreads();
  for (int st = 128; st >= C; st >>= 1) {
    if (tid < st) { s[tid] += s[tid + st]; q[tid] += q[tid + st]; }
    __syncthreads();
  }
  if (tid < C) {
    atomicAdd(&sum[c], s[tid]);
    atomicAdd(&sumsq[c], q[tid]);
  }
}

__global__ void bn_finalize(const float* __restrict__ sum,
                            const float* __restrict__ sumsq,
                            const float* __restrict__ g,
                            const float* __restrict__ be,
                            float* __restrict__ scale, float* __restrict__ shift,
                            int C, float invN) {
  int c = threadIdx.x;
  if (c >= C) return;
  float mean = sum[c] * invN;
  float var  = sumsq[c] * invN - mean * mean;
  float sc   = g[c] * rsqrtf(var + 1e-5f);
  scale[c] = sc;
  shift[c] = be[c] - mean * sc;
}

// ---------------------------------------------------------------------------
// Fused: BN apply + LeakyReLU + (optional) next-mask premultiply + f16
// convert. 4 channels per thread (v4f load -> half4 store). Padded channels
// [C, Cp) are written as zero.
// ---------------------------------------------------------------------------
__global__ __launch_bounds__(256) void bn_apply(
    const float* __restrict__ y, const float* __restrict__ scale,
    const float* __restrict__ shift, const float* __restrict__ mnext,
    _Float16* __restrict__ act, int C, int Cp, int useMask) {
  size_t e = ((size_t)blockIdx.x * blockDim.x + threadIdx.x) * 4;  // grid exact
  int c0 = (int)(e & (size_t)(Cp - 1));
  size_t pix = e / Cp;
  v4f yv = *(const v4f*)(y + e);
  float mval = useMask ? mnext[pix] : 1.0f;
  half4 hv;
#pragma unroll
  for (int r = 0; r < 4; ++r) {
    int c = c0 + r;
    float z = 0.0f;
    if (c < C) {
      z = scale[c] * yv[r] + shift[c];
      z = (z >= 0.0f) ? z : 0.01f * z;
      z *= mval;
    }
    hv[r] = (_Float16)z;
  }
  *(half4*)(act + e) = hv;
}

// ---------------------------------------------------------------------------
// Final 1x1 conv 128->64 via WMMA; LDS-staged weights; stores NCHW f32 with
// 2x b128 per lane (pixels are contiguous per lane in the D layout).
// ---------------------------------------------------------------------------
__global__ __launch_bounds__(256) void conv1x1_wmma(
    const _Float16* __restrict__ act, const _Float16* __restrict__ wp,
    const float* __restrict__ bias, float* __restrict__ out) {
  __shared__ alignas(32) _Float16 wlds[4 * 512];

  const int og    = blockIdx.x & 3;
  const int tile8 = blockIdx.x >> 2;

  {  // 2048 halfs = 256 threads * 16B, single pass
    const int idx    = threadIdx.x * 8;
    const int blk    = idx >> 9;
    const int within = idx & 511;
    *(half8*)&wlds[idx] =
        *(const half8*)(wp + ((size_t)((blk << 2) + og) << 9) + within);
  }
  __syncthreads();

  const int lane    = threadIdx.x & 31;
  const int pixbase = (tile8 << 7) + ((threadIdx.x >> 5) << 4);
  const int pl = lane & 15, hi = lane >> 4;

  const _Float16* ap = act + (size_t)(pixbase + pl) * 128;
  v8f acc = {0.f, 0.f, 0.f, 0.f, 0.f, 0.f, 0.f, 0.f};
#pragma unroll
  for (int kc = 0; kc < 4; ++kc) {
    const int c0 = (kc << 5) + (hi << 4);
    V16U a;
    a.h[0] = *(const half8*)(ap + c0);
    a.h[1] = *(const half8*)(ap + c0 + 8);
    const v16h bv = *(const v16h*)&wlds[(kc << 9) + (lane << 4)];
    acc = wmma_f16(a.v, bv, acc);
  }
  const float bc = bias[(og << 4) + pl];
  const int b = pixbase >> 16, yx = pixbase & (HW - 1);
  float* op = out + ((size_t)b * 64 + (og << 4) + pl) * HW + yx + (hi << 3);
  v4f o0, o1;
#pragma unroll
  for (int r = 0; r < 4; ++r) {
    o0[r] = acc[r] + bc;
    o1[r] = acc[4 + r] + bc;
  }
  *(v4f*)op       = o0;
  *(v4f*)(op + 4) = o1;
}

// ---------------------------------------------------------------------------
// Host orchestration (graph-capture safe: only kernel launches + memsetAsync).
// ---------------------------------------------------------------------------
extern "C" void kernel_launch(void* const* d_in, const int* in_sizes, int n_in,
                              void* d_out, int out_size, void* d_ws,
                              size_t ws_size, hipStream_t stream) {
  (void)in_sizes; (void)n_in; (void)out_size; (void)ws_size;

  const float* feat  = (const float*)d_in[0];
  const float* mask0 = (const float*)d_in[1];
  const float *W[5], *bb[5], *gg[5], *be[5];
  for (int l = 0; l < 5; ++l) {
    W[l]  = (const float*)d_in[2 + 4 * l];
    bb[l] = (const float*)d_in[3 + 4 * l];
    gg[l] = (const float*)d_in[4 + 4 * l];
    be[l] = (const float*)d_in[5 + 4 * l];
  }
  const float* Wf = (const float*)d_in[22];
  const float* bf = (const float*)d_in[23];

  static const int cinReal[5] = {32, 16, 32, 64, 64};
  static const int CpIn[5]    = {32, 32, 32, 64, 64};   // padded to mult of 32
  static const int Cout[5]    = {16, 32, 64, 64, 128};
  static const int CpOut[5]   = {32, 32, 64, 64, 128};
  static const int dil[5]     = {1, 2, 1, 2, 1};

  // Workspace carve-out
  char* ws = (char*)d_ws;
  size_t off = 0;
  auto carve = [&](size_t sz) -> char* {
    char* p = ws + off;
    off = (off + sz + 255) & ~(size_t)255;
    return p;
  };

  _Float16* wpack[5];
  int nkc[5], nocg[5];
  for (int l = 0; l < 5; ++l) {
    nkc[l]  = CpIn[l] / 32;
    nocg[l] = Cout[l] / 16;
    wpack[l] =
        (_Float16*)carve((size_t)9 * nkc[l] * nocg[l] * 512 * sizeof(_Float16));
  }
  _Float16* wpackF = (_Float16*)carve((size_t)16 * 512 * sizeof(_Float16));
  float* stats      = (float*)carve(5 * 2 * 128 * sizeof(float));
  float* scaleshift = (float*)carve(2 * 128 * sizeof(float));
  float* boxsum     = (float*)carve((size_t)NPIX * sizeof(float));
  float* maskA      = (float*)carve((size_t)NPIX * sizeof(float));
  float* maskB      = (float*)carve((size_t)NPIX * sizeof(float));
  float* ybuf       = (float*)carve((size_t)NPIX * 128 * sizeof(float));
  _Float16* actA    = (_Float16*)carve((size_t)NPIX * 128 * sizeof(_Float16));
  _Float16* actB    = (_Float16*)carve((size_t)NPIX * 128 * sizeof(_Float16));

  hipMemsetAsync(stats, 0, 5 * 2 * 128 * sizeof(float), stream);

  // Weight packing (tiny)
  for (int l = 0; l < 5; ++l) {
    int total = 9 * nkc[l] * nocg[l] * 512;
    pack_w<<<(total + 255) / 256, 256, 0, stream>>>(W[l], wpack[l], cinReal[l],
                                                    nkc[l], nocg[l], 9);
  }
  pack_w<<<(16 * 512 + 255) / 256, 256, 0, stream>>>(Wf, wpackF, 128, 4, 4, 1);

  // feat * mask -> f16 NHWC
  prep_feat<<<(unsigned)(((size_t)NPIX * 32) / 256), 256, 0, stream>>>(
      feat, mask0, actA);

  const float* mcur = mask0;
  _Float16* aIn  = actA;
  _Float16* aOut = actB;
  float* mbufs[2] = {maskA, maskB};
  float* maskOut = (float*)d_out + (size_t)8 * 64 * HW;  // tail of d_out

  for (int l = 0; l < 5; ++l) {
    float* mnext = (l == 4) ? maskOut : mbufs[l & 1];
    mask_step<<<NPIX / 256, 256, 0, stream>>>(mcur, boxsum, mnext, dil[l]);

    // block = 8 waves = 128 pixels x one 16-outch group
    int blocks = (NPIX / 128) * nocg[l];
    if (nkc[l] == 1) {
      conv3x3_sparse_wmma<1><<<blocks, 256, 0, stream>>>(
          aIn, wpack[l], boxsum, bb[l], ybuf, CpIn[l], nocg[l], CpOut[l],
          dil[l], (float)cinReal[l]);
    } else {
      conv3x3_sparse_wmma<2><<<blocks, 256, 0, stream>>>(
          aIn, wpack[l], boxsum, bb[l], ybuf, CpIn[l], nocg[l], CpOut[l],
          dil[l], (float)cinReal[l]);
    }

    float* sum   = stats + l * 256;
    float* sumsq = sum + 128;
    bn_stats<<<1024, 256, 0, stream>>>(ybuf, sum, sumsq, Cout[l], CpOut[l]);
    bn_finalize<<<1, 128, 0, stream>>>(sum, sumsq, gg[l], be[l], scaleshift,
                                       scaleshift + 128, Cout[l],
                                       1.0f / (float)NPIX);
    size_t total = (size_t)NPIX * CpOut[l];
    bn_apply<<<(unsigned)(total / 1024), 256, 0, stream>>>(
        ybuf, scaleshift, scaleshift + 128, mnext, aOut, Cout[l], CpOut[l],
        (l < 4) ? 1 : 0);

    mcur = mnext;
    _Float16* t = aIn; aIn = aOut; aOut = t;
  }

  // Final 1x1 conv -> NCHW f32 output
  conv1x1_wmma<<<(NPIX / 128) * 4, 256, 0, stream>>>(aIn, wpackF, bf,
                                                     (float*)d_out);
}
